// QQwen3Attention_79912161509834
// MI455X (gfx1250) — 16-bit and 8-bit matrix operand order repaired and checked by reading, compiled, not run on hardware
//
#include <hip/hip_runtime.h>
#include <cstdint>

typedef __attribute__((ext_vector_type(16))) _Float16 v16h;
typedef __attribute__((ext_vector_type(8)))  _Float16 v8h;
typedef __attribute__((ext_vector_type(8)))  float    v8f;
typedef __attribute__((ext_vector_type(8)))  int      v8i;

constexpr int Bc   = 2;
constexpr int Sc   = 1024;
constexpr int HIDc = 4096;
constexpr int NHc  = 32;
constexpr int NKVc = 8;
constexpr int HDc  = 128;
constexpr int SELc = 128;
constexpr int GSc  = 128;
constexpr int LOWc = HIDc - SELc;
constexpr int NGc  = LOWc / GSc;
constexpr int BSc  = Bc * Sc;
constexpr int RQKV = NHc*HDc + 2*NKVc*HDc;
constexpr float NEGF = -1e30f;

static __device__ __forceinline__ v8f zf8(){ v8f z = {0.f,0.f,0.f,0.f,0.f,0.f,0.f,0.f}; return z; }
static __device__ __forceinline__ v8i zi8(){ v8i z = {0,0,0,0,0,0,0,0}; return z; }
#define CAT16(a,b) __builtin_shufflevector(a, b, 0,1,2,3,4,5,6,7,8,9,10,11,12,13,14,15)

__global__ void k_quant_x(const float* __restrict__ x, const int* __restrict__ perm,
                          int8_t* __restrict__ xq, float* __restrict__ xs,
                          _Float16* __restrict__ xk)
{
    const int row = blockIdx.x;
    const int t   = threadIdx.x;
    __shared__ float red[256];
    const float* xr = x + (size_t)row * HIDc;

    float amax = 0.f;
    for (int j = t; j < LOWc; j += 256)
        amax = fmaxf(amax, fabsf(xr[perm[j]]));
    red[t] = amax; __syncthreads();
    for (int st = 128; st > 0; st >>= 1) {
        if (t < st) red[t] = fmaxf(red[t], red[t + st]);
        __syncthreads();
    }
    const float scale = fmaxf(red[0], 1e-5f) / 127.f;
    const float inv   = 1.f / scale;
    for (int j = t; j < LOWc; j += 256) {
        float q = fminf(fmaxf(rintf(xr[perm[j]] * inv), -128.f), 127.f);
        xq[(size_t)row * LOWc + j] = (int8_t)q;
    }
    if (t < SELc) xk[(size_t)row * SELc + t] = (_Float16)xr[perm[LOWc + t]];
    if (t == 0)   xs[row] = scale;
}

__global__ void k_quant_w(const float* __restrict__ W, const int* __restrict__ perm,
                          int8_t* __restrict__ wq, float* __restrict__ ws,
                          _Float16* __restrict__ wk, int row_off)
{
    const int r = blockIdx.x, g = blockIdx.y, t = threadIdx.x;
    const float* wr = W + (size_t)r * HIDc;
    if (g == NGc) {
        wk[(size_t)(row_off + r) * SELc + t] = (_Float16)wr[perm[LOWc + t]];
        return;
    }
    __shared__ float mn[128], mx[128];
    const int col = g * GSc + t;
    const float v = wr[perm[col]];
    mn[t] = v; mx[t] = v; __syncthreads();
    for (int st = 64; st > 0; st >>= 1) {
        if (t < st) { mn[t] = fminf(mn[t], mn[t+st]); mx[t] = fmaxf(mx[t], mx[t+st]); }
        __syncthreads();
    }
    const float scale = fmaxf(mx[0] - mn[0], 1e-5f) / 15.f;
    const float base  = fminf(fmaxf(rintf(-mn[0] / scale), 0.f), 15.f);
    const float q     = fminf(fmaxf(rintf(v / scale) + base, 0.f), 15.f) - base;
    wq[(size_t)(row_off + r) * LOWc + col] = (int8_t)q;
    if (t == 0) ws[(size_t)(row_off + r) * NGc + g] = scale;
}

__global__ void k_gemm(const int8_t* __restrict__ xq, const float* __restrict__ xs,
                       const _Float16* __restrict__ xk,
                       const int8_t* __restrict__ wq, const float* __restrict__ ws,
                       const _Float16* __restrict__ wk,
                       float* __restrict__ y, int R)
{
    const int wave = threadIdx.x >> 5, lane = threadIdx.x & 31;
    const int hi16 = lane >> 4, l16 = lane & 15;
    const int wm = wave >> 1, wn = wave & 1;
    const int m0 = blockIdx.y * 64 + wm * 16;
    const int n0 = blockIdx.x * 32 + wn * 16;
    const int am = m0 + l16;
    const int bn = n0 + l16;

    const int8_t* ap = xq + (size_t)am * LOWc;
    const int8_t* bp = wq + (size_t)bn * LOWc;

    float accf[8];
    #pragma unroll
    for (int r = 0; r < 8; r++) accf[r] = 0.f;

    for (int g = 0; g < NGc; g++) {
        v8i ci = zi8();
        #pragma unroll
        for (int kk = 0; kk < 2; kk++) {
            const int k0 = g * GSc + kk * 64;
            const int8_t* a8 = ap + k0 + hi16 * 8;
            int2 t0 = *(const int2*)(a8);
            int2 t1 = *(const int2*)(a8 + 16);
            int2 t2 = *(const int2*)(a8 + 32);
            int2 t3 = *(const int2*)(a8 + 48);
            v8i A; A[0]=t0.x; A[1]=t0.y; A[2]=t1.x; A[3]=t1.y;
                   A[4]=t2.x; A[5]=t2.y; A[6]=t3.x; A[7]=t3.y;
            const int8_t* b8 = bp + k0 + hi16 * 8;
            int2 u0 = *(const int2*)(b8);
            int2 u1 = *(const int2*)(b8 + 16);
            int2 u2 = *(const int2*)(b8 + 32);
            int2 u3 = *(const int2*)(b8 + 48);
            v8i Bv; Bv[0]=u0.x; Bv[1]=u0.y; Bv[2]=u1.x; Bv[3]=u1.y;
                    Bv[4]=u2.x; Bv[5]=u2.y; Bv[6]=u3.x; Bv[7]=u3.y;
            ci = __builtin_amdgcn_wmma_i32_16x16x64_iu8(true, A, true, Bv, ci, false, false);
        }
        const float wsg = ws[(size_t)bn * NGc + g];
        #pragma unroll
        for (int r = 0; r < 8; r++) accf[r] += (float)ci[r] * wsg;
    }

    v8f acck = zf8();
    const _Float16* akp = xk + (size_t)am * SELc;
    const _Float16* bkp = wk + (size_t)bn * SELc;
    #pragma unroll
    for (int c = 0; c < 4; c++) {
        v8h a0 = *(const v8h*)(akp + c * 32 + hi16 * 8);
        v8h a1 = *(const v8h*)(akp + c * 32 + hi16 * 8 + 16);
        v16h A = CAT16(a0, a1);
        v8h b0 = *(const v8h*)(bkp + c * 32 + hi16 * 8);
        v8h b1 = *(const v8h*)(bkp + c * 32 + hi16 * 8 + 16);
        v16h Bv = CAT16(b0, b1);
        acck = __builtin_amdgcn_wmma_f32_16x16x32_f16(false, A, false, Bv,
                                                      (short)0, acck, false, false);
    }

    #pragma unroll
    for (int r = 0; r < 8; r++) {
        const int m = m0 + r + hi16 * 8;
        y[(size_t)m * R + n0 + l16] = accf[r] * xs[m] + acck[r];
    }
}

__global__ void k_postproc(const float* __restrict__ y,
                           const float* __restrict__ cosb, const float* __restrict__ sinb,
                           const float* __restrict__ qnw,  const float* __restrict__ knw,
                           _Float16* __restrict__ qf, _Float16* __restrict__ kf,
                           _Float16* __restrict__ vtf)
{
    const int row = blockIdx.x;
    const int slot = blockIdx.y;
    const int d = threadIdx.x;
    const int b = row / Sc, s = row % Sc;
    __shared__ float sv[HDc];
    __shared__ float r1[HDc], r2[HDc];

    float val; int h;
    if (slot < 32)      { h = slot;      val = y[(size_t)row * RQKV + h * HDc + d]; }
    else if (slot < 40) { h = slot - 32; val = y[(size_t)row * RQKV + 4096 + h * HDc + d]; }
    else                { h = slot - 40; val = y[(size_t)row * RQKV + 5120 + h * HDc + d]; }

    if (slot < 40) {
        r1[d] = val * val; __syncthreads();
        for (int st = 64; st > 0; st >>= 1) { if (d < st) r1[d] += r1[d + st]; __syncthreads(); }
        const float rms = rsqrtf(r1[0] * (1.f / HDc) + 1e-6f);
        __syncthreads();
        const float w = (slot < 32) ? qnw[d] : knw[d];
        val = val * rms * w;
    }
    if (slot >= 32) {
        r1[d] = val; r2[d] = val; __syncthreads();
        for (int st = 64; st > 0; st >>= 1) {
            if (d < st) { r1[d] = fminf(r1[d], r1[d+st]); r2[d] = fmaxf(r2[d], r2[d+st]); }
            __syncthreads();
        }
        const float scale = fmaxf(r2[0] - r1[0], 1e-5f) / 15.f;
        const float base  = fminf(fmaxf(rintf(-r1[0] / scale), 0.f), 15.f);
        __syncthreads();
        const float q = fminf(fmaxf(rintf(val / scale) + base, 0.f), 15.f) - base;
        val = q * scale;
    }
    if (slot < 40) {
        sv[d] = val; __syncthreads();
        const float c  = cosb[(size_t)row * HDc + d];
        const float sn = sinb[(size_t)row * HDc + d];
        const float rot = (d < 64) ? -sv[d + 64] : sv[d - 64];
        val = val * c + rot * sn;
    }
    if (slot < 32)
        qf[(((size_t)(b * NHc + h)) * Sc + s) * HDc + d] = (_Float16)val;
    else if (slot < 40)
        kf[(((size_t)(b * NKVc + h)) * Sc + s) * HDc + d] = (_Float16)val;
    else
        vtf[(((size_t)(b * NKVc + h)) * HDc + d) * Sc + s] = (_Float16)val;
}

__global__ void k_attn(const _Float16* __restrict__ qf, const _Float16* __restrict__ kf,
                       const _Float16* __restrict__ vtf, float* __restrict__ outp)
{
    const int b = blockIdx.z, h = blockIdx.y;
    const int wave = threadIdx.x >> 5, lane = threadIdx.x & 31;
    const int hi16 = lane >> 4, l16 = lane & 15;
    const int qt0 = blockIdx.x * 128 + wave * 16;
    const int hkv = h >> 2;
    __shared__ _Float16 lp[8][16 * 32];

    const _Float16* qbase = qf  + ((size_t)(b * NHc  + h  )) * Sc * HDc;
    const _Float16* kbase = kf  + ((size_t)(b * NKVc + hkv)) * Sc * HDc;
    const _Float16* vbase = vtf + ((size_t)(b * NKVc + hkv)) * HDc * Sc;

    const float qscl = 0.08838834764831845f;
    v16h qA[4];
    {
        const _Float16* qp = qbase + (size_t)(qt0 + l16) * HDc;
        #pragma unroll
        for (int c = 0; c < 4; c++) {
            v8h a0 = *(const v8h*)(qp + c * 32 + hi16 * 8);
            v8h a1 = *(const v8h*)(qp + c * 32 + hi16 * 8 + 16);
            v16h qv = CAT16(a0, a1);
            #pragma unroll
            for (int i = 0; i < 16; i++)
                qA[c][i] = (_Float16)((float)qv[i] * qscl);
        }
    }

    float mi[8], li[8];
    v8f o[8];
    #pragma unroll
    for (int r = 0; r < 8; r++) { mi[r] = -3e38f; li[r] = 0.f; o[r] = zf8(); }

    const int nj = (qt0 + 47) >> 5;
    for (int j = 0; j < nj; j++) {
        const int jb = j * 32;
        float p[2][8], rm[8], alpha[8], rs[8];
        #pragma unroll
        for (int r = 0; r < 8; r++) rm[r] = -3e38f;

        #pragma unroll
        for (int t = 0; t < 2; t++) {
            v8f sc = zf8();
            const int key = jb + t * 16 + l16;
            const _Float16* kp = kbase + (size_t)key * HDc;
            #pragma unroll
            for (int c = 0; c < 4; c++) {
                const int d0 = c * 32 + hi16 * 8;
                v8h b0 = *(const v8h*)(kp + d0);
                v8h b1 = *(const v8h*)(kp + d0 + 16);
                v16h Bv = CAT16(b0, b1);
                sc = __builtin_amdgcn_wmma_f32_16x16x32_f16(false, qA[c], false, Bv,
                                                            (short)0, sc, false, false);
            }
            #pragma unroll
            for (int r = 0; r < 8; r++) {
                const int qg = qt0 + r + hi16 * 8;
                const float v = (key > qg) ? NEGF : sc[r];
                p[t][r] = v;
                rm[r] = fmaxf(rm[r], v);
            }
        }
        #pragma unroll
        for (int m = 1; m < 16; m <<= 1)
            #pragma unroll
            for (int r = 0; r < 8; r++) rm[r] = fmaxf(rm[r], __shfl_xor(rm[r], m, 32));
        #pragma unroll
        for (int r = 0; r < 8; r++) {
            const float mn = fmaxf(mi[r], rm[r]);
            alpha[r] = __expf(mi[r] - mn);
            mi[r] = mn;
            p[0][r] = __expf(p[0][r] - mn);
            p[1][r] = __expf(p[1][r] - mn);
            rs[r] = p[0][r] + p[1][r];
        }
        #pragma unroll
        for (int m = 1; m < 16; m <<= 1)
            #pragma unroll
            for (int r = 0; r < 8; r++) rs[r] += __shfl_xor(rs[r], m, 32);
        #pragma unroll
        for (int r = 0; r < 8; r++) li[r] = li[r] * alpha[r] + rs[r];
        #pragma unroll
        for (int dt = 0; dt < 8; dt++)
            #pragma unroll
            for (int r = 0; r < 8; r++) o[dt][r] *= alpha[r];

        #pragma unroll
        for (int t = 0; t < 2; t++)
            #pragma unroll
            for (int r = 0; r < 8; r++)
                lp[wave][(r + hi16 * 8) * 32 + t * 16 + l16] = (_Float16)p[t][r];
        asm volatile("s_wait_dscnt 0" ::: "memory");
        v16h pA;
        {
            const _Float16* pp = &lp[wave][l16 * 32 + hi16 * 8];
            v8h a0 = *(const v8h*)pp;
            v8h a1 = *(const v8h*)(pp + 16);
            pA = CAT16(a0, a1);
        }
        #pragma unroll
        for (int dt = 0; dt < 8; dt++) {
            const int d = dt * 16 + l16;
            const _Float16* vp = vbase + (size_t)d * Sc + jb + hi16 * 8;
            v8h b0 = *(const v8h*)vp;
            v8h b1 = *(const v8h*)(vp + 16);
            v16h Bv = CAT16(b0, b1);
            o[dt] = __builtin_amdgcn_wmma_f32_16x16x32_f16(false, pA, false, Bv,
                                                           (short)0, o[dt], false, false);
        }
    }

    #pragma unroll
    for (int dt = 0; dt < 8; dt++) {
        #pragma unroll
        for (int r = 0; r < 8; r++) {
            const int srow = qt0 + r + hi16 * 8;
            outp[((size_t)(b * Sc + srow)) * (NHc * HDc) + h * HDc + dt * 16 + l16]
                = o[dt][r] / li[r];
        }
    }
}

extern "C" void kernel_launch(void* const* d_in, const int* in_sizes, int n_in,
                              void* d_out, int out_size, void* d_ws, size_t ws_size,
                              hipStream_t stream)
{
    (void)in_sizes; (void)n_in; (void)out_size; (void)ws_size;
    const float* hs   = (const float*)d_in[0];
    const float* cosb = (const float*)d_in[1];
    const float* sinb = (const float*)d_in[2];
    const float* Wq   = (const float*)d_in[3];
    const float* Wk   = (const float*)d_in[4];
    const float* Wv   = (const float*)d_in[5];
    const float* Wo   = (const float*)d_in[6];
    const float* qnw  = (const float*)d_in[7];
    const float* knw  = (const float*)d_in[8];
    const int*   pqkv = (const int*)d_in[9];
    const int*   po   = (const int*)d_in[10];

    char* ws = (char*)d_ws;
    size_t off = 0;
    auto carve = [&](size_t bytes) { char* p = ws + off; off = (off + bytes + 255) & ~(size_t)255; return p; };

    int8_t*   xq   = (int8_t*)  carve((size_t)BSc * LOWc);
    float*    xs   = (float*)   carve((size_t)BSc * 4);
    _Float16* xk   = (_Float16*)carve((size_t)BSc * SELc * 2);
    int8_t*   wq3  = (int8_t*)  carve((size_t)RQKV * LOWc);
    float*    ws3  = (float*)   carve((size_t)RQKV * NGc * 4);
    _Float16* wk3  = (_Float16*)carve((size_t)RQKV * SELc * 2);
    float*    yqkv = (float*)   carve((size_t)BSc * RQKV * 4);
    _Float16* qf   = (_Float16*)carve((size_t)Bc * NHc  * Sc * HDc * 2);
    _Float16* kf   = (_Float16*)carve((size_t)Bc * NKVc * Sc * HDc * 2);
    _Float16* vtf  = (_Float16*)carve((size_t)Bc * NKVc * HDc * Sc * 2);
    float*    ao   = (float*)   carve((size_t)BSc * (NHc * HDc) * 4);
    int8_t*   oq   = (int8_t*)  carve((size_t)BSc * LOWc);
    float*    osl  = (float*)   carve((size_t)BSc * 4);
    _Float16* okp  = (_Float16*)carve((size_t)BSc * SELc * 2);
    int8_t*   woq  = (int8_t*)  carve((size_t)HIDc * LOWc);
    float*    wos  = (float*)   carve((size_t)HIDc * NGc * 4);
    _Float16* wok  = (_Float16*)carve((size_t)HIDc * SELc * 2);

    k_quant_x<<<BSc, 256, 0, stream>>>(hs, pqkv, xq, xs, xk);
    k_quant_w<<<dim3(NHc * HDc,  NGc + 1), 128, 0, stream>>>(Wq, pqkv, wq3, ws3, wk3, 0);
    k_quant_w<<<dim3(NKVc * HDc, NGc + 1), 128, 0, stream>>>(Wk, pqkv, wq3, ws3, wk3, NHc * HDc);
    k_quant_w<<<dim3(NKVc * HDc, NGc + 1), 128, 0, stream>>>(Wv, pqkv, wq3, ws3, wk3, NHc * HDc + NKVc * HDc);
    k_gemm<<<dim3(RQKV / 32, BSc / 64), 256, 0, stream>>>(xq, xs, xk, wq3, ws3, wk3, yqkv, RQKV);
    k_postproc<<<dim3(BSc, NHc + 2 * NKVc), 128, 0, stream>>>(yqkv, cosb, sinb, qnw, knw, qf, kf, vtf);
    k_attn<<<dim3(Sc / 128, NHc, Bc), 256, 0, stream>>>(qf, kf, vtf, ao);
    k_quant_x<<<BSc, 256, 0, stream>>>(ao, po, oq, osl, okp);
    k_quant_w<<<dim3(HIDc, NGc + 1), 128, 0, stream>>>(Wo, po, woq, wos, wok, 0);
    k_gemm<<<dim3(HIDc / 32, BSc / 64), 256, 0, stream>>>(oq, osl, okp, woq, wos, wok, (float*)d_out, HIDc);
}
